// Makser_27255862460899
// MI455X (gfx1250) — matrix operand order repaired and checked by reading, not compiled, not run on hardware
//
#include <hip/hip_runtime.h>
#include <hip/hip_bf16.h>

typedef __attribute__((ext_vector_type(16))) __bf16 v16bf;
typedef __attribute__((ext_vector_type(8)))  float  v8f;

#define DD 128

// ---------------------------------------------------------------------------
// h[v][c] = node_emb1[x[v,0]][c] + node_emb2[x[v,1]][c]
__global__ void k_init_embed(const int* __restrict__ x,
                             const float* __restrict__ ne1,
                             const float* __restrict__ ne2,
                             float* __restrict__ h, int total)
{
    int idx = blockIdx.x * blockDim.x + threadIdx.x;
    if (idx >= total) return;
    int v = idx >> 7, c = idx & 127;
    int a0 = x[2 * v], a1 = x[2 * v + 1];
    h[idx] = ne1[(size_t)a0 * DD + c] + ne2[(size_t)a1 * DD + c];
}

// agg[v][c] = h[v][c] + selfloop_emb1[c] + selfloop_emb2[c]
__global__ void k_selfloop(const float* __restrict__ h,
                           const float* __restrict__ se1,
                           const float* __restrict__ se2,
                           float* __restrict__ agg, int total)
{
    int idx = blockIdx.x * blockDim.x + threadIdx.x;
    if (idx >= total) return;
    int c = idx & 127;
    agg[idx] = h[idx] + se1[c] + se2[c];
}

// one wave per edge: agg[dst] += h[src] + e1[ea0] + e2[ea1]   (4 floats/lane)
__global__ void k_edge_scatter(const int* __restrict__ src,
                               const int* __restrict__ dst,
                               const int* __restrict__ ea,
                               const float* __restrict__ h,
                               const float* __restrict__ e1,
                               const float* __restrict__ e2,
                               float* __restrict__ agg, int E)
{
    int wave = (int)((blockIdx.x * (unsigned)blockDim.x + threadIdx.x) >> 5);
    int lane = threadIdx.x & 31;
    if (wave >= E) return;
    int s = src[wave], d = dst[wave];
    int a0 = ea[2 * wave], a1 = ea[2 * wave + 1];
    int c = lane * 4;
    float4 m  = *(const float4*)(h  + (size_t)s  * DD + c);
    float4 t1 = *(const float4*)(e1 + (size_t)a0 * DD + c);
    float4 t2 = *(const float4*)(e2 + (size_t)a1 * DD + c);
    float* ap = agg + (size_t)d * DD + c;
    atomicAdd(ap + 0, m.x + t1.x + t2.x);
    atomicAdd(ap + 1, m.y + t1.y + t2.y);
    atomicAdd(ap + 2, m.z + t1.z + t2.z);
    atomicAdd(ap + 3, m.w + t1.w + t2.w);
}

// Wt[n*K + k] = (bf16) W[k*Ncols + n]   (transpose + convert; weights are tiny)
__global__ void k_wt_prep(const float* __restrict__ W, __bf16* __restrict__ Wt,
                          int K, int Ncols)
{
    int idx = blockIdx.x * blockDim.x + threadIdx.x;
    if (idx >= K * Ncols) return;
    int n = idx / K, k = idx - n * K;
    Wt[idx] = (__bf16)W[(size_t)k * Ncols + n];
}

__global__ void k_zero(float* __restrict__ p, int n)
{
    int idx = blockIdx.x * blockDim.x + threadIdx.x;
    if (idx < n) p[idx] = 0.f;
}

// column sums / sums-of-squares -> stats[0..127]=sum, stats[128..255]=sumsq
__global__ void k_bn_stats(const float* __restrict__ hh, float* __restrict__ stats,
                           int Nrows)
{
    int c = threadIdx.x;                 // blockDim = 128
    float s = 0.f, q = 0.f;
    for (int r = blockIdx.x; r < Nrows; r += gridDim.x) {
        float v = hh[(size_t)r * DD + c];
        s += v; q += v * v;
    }
    atomicAdd(&stats[c], s);
    atomicAdd(&stats[DD + c], q);
}

__global__ void k_bn_apply(const float* __restrict__ hh, const float* __restrict__ stats,
                           const float* __restrict__ g, const float* __restrict__ b,
                           float* __restrict__ hout, int total, int doRelu, float invN)
{
    int idx = blockIdx.x * blockDim.x + threadIdx.x;
    if (idx >= total) return;
    int c = idx & 127;
    float mean = stats[c] * invN;
    float var  = stats[DD + c] * invN - mean * mean;
    float sc   = g[c] * rsqrtf(var + 1e-5f);
    float v = (hh[idx] - mean) * sc + b[c];
    if (doRelu) v = fmaxf(v, 0.f);
    hout[idx] = v;
}

// ---------------------------------------------------------------------------
// C[M x Ncols] = act(A[M x KK] @ W + bias); W pre-transposed bf16 [Ncols x KK].
// One wave computes a 16x64 tile: 4 accumulators, v_wmma_f32_16x16x32_bf16.
// Templated on K / input / output precision: constant trip count, no hot-loop
// branches; all 4 B fragments are loaded before the 4 WMMAs so vmem overlaps
// matrix issue (waits become loadcnt<=3/2/1/0 instead of 0 each time).
template <int KK, int ABF16, int OUTBF16>
__global__ void gemm_bf16_wmma(const void* __restrict__ Av,
                               const __bf16* __restrict__ Wt,
                               const float* __restrict__ bias,
                               void* __restrict__ Ov, int doRelu,
                               int M, int Ncols)
{
    const int wave = (int)((blockIdx.x * (unsigned)blockDim.x + threadIdx.x) >> 5);
    const int lane = threadIdx.x & 31;
    const int ntN  = Ncols >> 6;               // 64-col tiles per row-tile
    const int tileM = wave / ntN;
    const int colBase = (wave - tileM * ntN) << 6;
    const int rowBase = tileM << 4;
    if (rowBase >= M) return;                  // wave-uniform: EXEC stays full

    int rm = rowBase + (lane & 15);
    if (rm >= M) rm = M - 1;                   // clamp loads; stores guarded
    const int hiA = (lane & 16) ? 8 : 0;       // A: K split per ISA layout
    const int hiB = (lane & 16) ? 8 : 0;       // B: K split per ISA layout
    const int colL = colBase + (lane & 15);

    v8f acc0 = {}, acc1 = {}, acc2 = {}, acc3 = {};

    const float*  Af = (const float*)Av;
    const __bf16* Ab = (const __bf16*)Av;
    union U { v16bf v; uint4 q[2]; };

    #pragma unroll
    for (int k = 0; k < KK; k += 32) {
        // ---- all 4 B fragments first (2x16B per lane: K [k+hiB,+8) and [k+hiB+16,+8)) ----
        const __bf16* bp = Wt + (size_t)colL * KK + (k + hiB);
        U b0, b1, b2, b3;
        b0.q[0] = *(const uint4*)(bp);               b0.q[1] = *(const uint4*)(bp + 16);
        b1.q[0] = *(const uint4*)(bp + 16 * KK);     b1.q[1] = *(const uint4*)(bp + 16 * KK + 16);
        b2.q[0] = *(const uint4*)(bp + 32 * KK);     b2.q[1] = *(const uint4*)(bp + 32 * KK + 16);
        b3.q[0] = *(const uint4*)(bp + 48 * KK);     b3.q[1] = *(const uint4*)(bp + 48 * KK + 16);

        // ---- A fragment ----
        v16bf a;
        if (ABF16) {
            const __bf16* ap = Ab + (size_t)rm * KK + (k + hiA);
            U ua;
            ua.q[0] = *(const uint4*)(ap);           // K [k+hiA,    +8)
            ua.q[1] = *(const uint4*)(ap + 16);      // K [k+hiA+16, +8)
            a = ua.v;
        } else {
            const float* ap = Af + (size_t)rm * KK + (k + hiA);
            float4 f0 = *(const float4*)(ap);
            float4 f1 = *(const float4*)(ap + 4);
            float4 f2 = *(const float4*)(ap + 16);
            float4 f3 = *(const float4*)(ap + 20);
            a[0]=(__bf16)f0.x; a[1]=(__bf16)f0.y; a[2]=(__bf16)f0.z; a[3]=(__bf16)f0.w;
            a[4]=(__bf16)f1.x; a[5]=(__bf16)f1.y; a[6]=(__bf16)f1.z; a[7]=(__bf16)f1.w;
            a[8]=(__bf16)f2.x; a[9]=(__bf16)f2.y; a[10]=(__bf16)f2.z; a[11]=(__bf16)f2.w;
            a[12]=(__bf16)f3.x; a[13]=(__bf16)f3.y; a[14]=(__bf16)f3.z; a[15]=(__bf16)f3.w;
        }

        acc0 = __builtin_amdgcn_wmma_f32_16x16x32_bf16(false, a, false, b0.v, (short)0, acc0, false, false);
        acc1 = __builtin_amdgcn_wmma_f32_16x16x32_bf16(false, a, false, b1.v, (short)0, acc1, false, false);
        acc2 = __builtin_amdgcn_wmma_f32_16x16x32_bf16(false, a, false, b2.v, (short)0, acc2, false, false);
        acc3 = __builtin_amdgcn_wmma_f32_16x16x32_bf16(false, a, false, b3.v, (short)0, acc3, false, false);
    }

    // C layout: lane%16 = column, lanes>=16 hold rows +8; VGPR r = row +r
    const int c0 = colBase + (lane & 15);
    const float bb0 = bias[c0], bb1 = bias[c0 + 16], bb2 = bias[c0 + 32], bb3 = bias[c0 + 48];
    const int rowTop = rowBase + ((lane & 16) ? 8 : 0);
    float*  Of = (float*)Ov;
    __bf16* Ob = (__bf16*)Ov;
    #pragma unroll
    for (int r = 0; r < 8; ++r) {
        int row = rowTop + r;
        if (row < M) {
            size_t base = (size_t)row * Ncols + c0;
            float v0 = acc0[r] + bb0, v1 = acc1[r] + bb1;
            float v2 = acc2[r] + bb2, v3 = acc3[r] + bb3;
            if (doRelu) {
                v0 = fmaxf(v0, 0.f); v1 = fmaxf(v1, 0.f);
                v2 = fmaxf(v2, 0.f); v3 = fmaxf(v3, 0.f);
            }
            if (OUTBF16) {
                Ob[base] = (__bf16)v0; Ob[base + 16] = (__bf16)v1;
                Ob[base + 32] = (__bf16)v2; Ob[base + 48] = (__bf16)v3;
            } else {
                Of[base] = v0; Of[base + 16] = v1;
                Of[base + 32] = v2; Of[base + 48] = v3;
            }
        }
    }
}

// prob[row] = sigmoid(dot(h2[row], mW3) + mb3) ; one wave per row
__global__ void k_head_out(const float* __restrict__ h2, const float* __restrict__ w3,
                           const float* __restrict__ b3, float* __restrict__ out, int Nrows)
{
    int wave = (int)((blockIdx.x * (unsigned)blockDim.x + threadIdx.x) >> 5);
    int lane = threadIdx.x & 31;
    if (wave >= Nrows) return;
    float4 a = *(const float4*)(h2 + (size_t)wave * DD + lane * 4);
    float4 b = *(const float4*)(w3 + lane * 4);
    float p = a.x * b.x + a.y * b.y + a.z * b.z + a.w * b.w;
    for (int off = 16; off > 0; off >>= 1) p += __shfl_xor(p, off, 32);
    if (lane == 0) out[wave] = 1.0f / (1.0f + expf(-(p + b3[0])));
}

// ---------------------------------------------------------------------------
extern "C" void kernel_launch(void* const* d_in, const int* in_sizes, int n_in,
                              void* d_out, int out_size, void* d_ws, size_t ws_size,
                              hipStream_t stream)
{
    const int*   x    = (const int*)d_in[0];
    const int*   ei   = (const int*)d_in[1];
    const int*   ea   = (const int*)d_in[2];
    const float* ne1  = (const float*)d_in[3];
    const float* ne2  = (const float*)d_in[4];
    const float* ee1  = (const float*)d_in[5];   // (L,6,128)
    const float* ee2  = (const float*)d_in[6];   // (L,3,128)
    const float* W1   = (const float*)d_in[7];   // (L,128,256)
    const float* b1   = (const float*)d_in[8];   // (L,256)
    const float* W2   = (const float*)d_in[9];   // (L,256,128)
    const float* b2   = (const float*)d_in[10];  // (L,128)
    const float* bn_g = (const float*)d_in[11];
    const float* bn_b = (const float*)d_in[12];
    const float* mW1  = (const float*)d_in[13];
    const float* mb1  = (const float*)d_in[14];
    const float* mW2  = (const float*)d_in[15];
    const float* mb2  = (const float*)d_in[16];
    const float* mW3  = (const float*)d_in[17];
    const float* mb3  = (const float*)d_in[18];

    const int Nn = in_sizes[0] / 2;
    const int Ee = in_sizes[1] / 2;
    const int Ll = in_sizes[8] / 256;
    const int* srcP = ei;
    const int* dstP = ei + Ee;

    // ---- workspace layout (256B-aligned) ----
    char* ws = (char*)d_ws;
    size_t off = 0;
    auto take = [&](size_t bytes) { size_t r = off; off = (off + bytes + 255) & ~(size_t)255; return r; };
    const size_t WSZ = 32768;                       // 128*256 elements per weight block
    __bf16* wt    = (__bf16*)(ws + take(12 * WSZ * sizeof(__bf16)));
    float*  stats = (float*) (ws + take(2 * DD * sizeof(float)));
    float*  h     = (float*) (ws + take((size_t)Nn * DD * sizeof(float)));
    float*  agg   = (float*) (ws + take((size_t)Nn * DD * sizeof(float)));   // also hh2
    __bf16* hh1   = (__bf16*)(ws + take((size_t)Nn * 2 * DD * sizeof(__bf16)));

    const int totND = Nn * DD;
    const dim3 blk(256);
    const int gridND = (totND + 255) / 256;

    // ---- weight prep: transpose + f32->bf16 (all layers + head) ----
    for (int i = 0; i < Ll; ++i) {
        k_wt_prep<<<dim3(128), blk, 0, stream>>>(W1 + (size_t)i * 128 * 256, wt + (2 * i) * WSZ, 128, 256);
        k_wt_prep<<<dim3(128), blk, 0, stream>>>(W2 + (size_t)i * 256 * 128, wt + (2 * i + 1) * WSZ, 256, 128);
    }
    k_wt_prep<<<dim3(128), blk, 0, stream>>>(mW1, wt + 10 * WSZ, 128, 256);
    k_wt_prep<<<dim3(128), blk, 0, stream>>>(mW2, wt + 11 * WSZ, 256, 128);

    // ---- initial node embedding ----
    k_init_embed<<<dim3(gridND), blk, 0, stream>>>(x, ne1, ne2, h, totND);

    // GEMM1: A f32 [N,128] -> out bf16 [N,256]
    auto launch_g1 = [&](const float* A, const __bf16* Wtp, const float* bias,
                         __bf16* O, int relu) {
        int waves = ((Nn + 15) / 16) * (256 >> 6);
        int blocks = (waves * 32 + 255) / 256;
        gemm_bf16_wmma<128, 0, 1><<<dim3(blocks), blk, 0, stream>>>(A, Wtp, bias, O, relu, Nn, 256);
    };
    // GEMM2: A bf16 [N,256] -> out f32 [N,128]
    auto launch_g2 = [&](const __bf16* A, const __bf16* Wtp, const float* bias,
                         float* O, int relu) {
        int waves = ((Nn + 15) / 16) * (128 >> 6);
        int blocks = (waves * 32 + 255) / 256;
        gemm_bf16_wmma<256, 1, 0><<<dim3(blocks), blk, 0, stream>>>(A, Wtp, bias, O, relu, Nn, 128);
    };

    // ---- GIN layers ----
    for (int i = 0; i < Ll; ++i) {
        const float* e1 = ee1 + (size_t)i * 6 * DD;
        const float* e2 = ee2 + (size_t)i * 3 * DD;
        // agg = self-loop message (h + emb1[4] + emb2[0])
        k_selfloop<<<dim3(gridND), blk, 0, stream>>>(h, e1 + 4 * DD, e2, agg, totND);
        // scatter-add real edges (one wave/edge; agg stays in the 192MB L2)
        int eBlocks = (Ee * 32 + 255) / 256;
        k_edge_scatter<<<dim3(eBlocks), blk, 0, stream>>>(srcP, dstP, ea, h, e1, e2, agg, Ee);
        // hh1 = relu(agg @ W1 + b1)   [bf16 out feeds next WMMA directly]
        launch_g1(agg, wt + (2 * i) * WSZ, b1 + (size_t)i * 256, hh1, 1);
        // hh2 = hh1 @ W2 + b2  (reuse agg buffer, f32)
        launch_g2(hh1, wt + (2 * i + 1) * WSZ, b2 + (size_t)i * 128, agg, 0);
        // batch-norm
        k_zero<<<dim3(1), blk, 0, stream>>>(stats, 2 * DD);
        k_bn_stats<<<dim3(1024), dim3(128), 0, stream>>>(agg, stats, Nn);
        k_bn_apply<<<dim3(gridND), blk, 0, stream>>>(agg, stats, bn_g + (size_t)i * DD,
                                                     bn_b + (size_t)i * DD, h, totND,
                                                     (i < Ll - 1) ? 1 : 0, 1.0f / (float)Nn);
    }

    // ---- MLP sigmoid head ----
    launch_g1(h, wt + 10 * WSZ, mb1, hh1, 1);      // relu(h@mW1+mb1)   -> bf16
    launch_g2(hh1, wt + 11 * WSZ, mb2, agg, 1);    // relu(...@mW2+mb2) -> f32
    int hBlocks = (Nn * 32 + 255) / 256;
    k_head_out<<<dim3(hBlocks), blk, 0, stream>>>(agg, mW3, mb3, (float*)d_out, Nn);
    (void)n_in; (void)out_size; (void)ws_size;
}
